// SATformer_27144193311176
// MI455X (gfx1250) — matrix operand order repaired and checked by reading, not compiled, not run on hardware
//
#include <hip/hip_runtime.h>
#include <hip/hip_bf16.h>
#include <stdint.h>

// ---------------- constants (match reference) ----------------
#define H        128
#define NLIT     131072
#define NCLAUSE  65536
#define E_EDGES  524288
#define NROUND   10
#define WWIN     16
#define LEVELS   4
#define NHEAD    8
#define DH       (H / NHEAD)     // 16

typedef __attribute__((ext_vector_type(16))) _Float16 v16h;
typedef __attribute__((ext_vector_type(8)))  _Float16 v8h;
typedef __attribute__((ext_vector_type(8)))  float    v8f;

// =====================================================================
//  Fused multi-segment WMMA GEMM.
//    C[N, Kout] = [relu]( concat_k(A0,A1,A2)[N, NSEG*128] @ W^T + bias )
//  W is f16 row-major [Kout, NSEG*128] (pre-concatenated along K).
//  A1 rows optionally remapped: row' = (row + shift1) % mod1  (lflip).
//  Block = 256 threads (8 waves). A tile (16 x NSEG*128) staged in LDS
//  once, reused across all Kout/128 column groups. Per column group all
//  B fragments are fetched up-front (one load clause), then the WMMA
//  chain runs with decaying load waits.
// =====================================================================
template<int NSEG>
__global__ __launch_bounds__(256)
void gemm_wmma_kernel(const float* __restrict__ A0,
                      const float* __restrict__ A1,
                      const float* __restrict__ A2,
                      int shift1, int mod1,
                      const _Float16* __restrict__ W,
                      const float* __restrict__ bias,
                      float* __restrict__ C, int ldc,
                      int Kout, int relu)
{
    constexpr int KT  = NSEG * 128;
    constexpr int NKT = NSEG * 4;
    __shared__ __align__(16) _Float16 As[16][KT + 8];   // pad vs LDS banks

    const int tid   = threadIdx.x;     // 0..255
    const int wave  = tid >> 5;        // 0..7
    const int lane  = tid & 31;        // wave32
    const int mBase = blockIdx.x * 16;

    // cooperative staging of the A tile segments, f32 -> f16 (float4 reads)
    const float* segs[3] = { A0, A1, A2 };
#pragma unroll
    for (int s = 0; s < NSEG; ++s) {
        const float* A = segs[s];
        for (int idx = tid; idx < 16 * 32; idx += 256) {
            int r  = idx >> 5;
            int c4 = idx & 31;
            int row = mBase + r;
            if (s == 1 && mod1) row = (row + shift1) % mod1;
            float4 v = ((const float4*)(A + (size_t)row * 128))[c4];
            _Float16* dst = &As[r][s * 128 + c4 * 4];
            dst[0] = (_Float16)v.x;
            dst[1] = (_Float16)v.y;
            dst[2] = (_Float16)v.z;
            dst[3] = (_Float16)v.w;
        }
    }
    __syncthreads();

    const int m  = lane & 15;          // A row within tile / C col index n
    const int hi = lane >> 4;
    const int n  = m;

    const int nCG = Kout >> 7;         // column groups of 128
    for (int cg = 0; cg < nCG; ++cg) {
        const int oBase = cg * 128 + wave * 16;
        const _Float16* wrow = W + (size_t)(oBase + n) * KT;
        __builtin_prefetch(wrow, 0, 1);          // global_prefetch_b8

        // bias for this lane's output column (unconditional; no branch in loop)
        const float bvv = bias[oBase + n];

        // fetch ALL B fragments for this column group up-front
        // (B[k][n] = W[oBase+n][koff+k]; lanes 0-15 hold K=koff+0..15,
        //  lanes 16-31 hold K=koff+16..31)
        v16h bfrag[NKT];
#pragma unroll
        for (int kt = 0; kt < NKT; ++kt) {
            const _Float16* bp = wrow + kt * 32 + hi * 16;
            v8h blo = *(const v8h*)(bp);
            v8h bhi = *(const v8h*)(bp + 8);
#pragma unroll
            for (int i = 0; i < 8; ++i) {
                bfrag[kt][i]     = blo[i];
                bfrag[kt][8 + i] = bhi[i];
            }
        }

        v8f acc;
#pragma unroll
        for (int r = 0; r < 8; ++r) acc[r] = 0.0f;

#pragma unroll
        for (int kt = 0; kt < NKT; ++kt) {
            const int koff = kt * 32;
            // A fragment (contiguous k order, matches B):
            //   a[0..7]  = A[m][koff + hi*16 .. +7]
            //   a[8..15] = A[m][koff + hi*16 + 8 .. +7]
            const _Float16* ap = &As[m][koff + hi * 16];
            v8h alo = *(const v8h*)(ap);
            v8h ahi = *(const v8h*)(ap + 8);
            v16h a;
#pragma unroll
            for (int i = 0; i < 8; ++i) { a[i] = alo[i]; a[8 + i] = ahi[i]; }

            acc = __builtin_amdgcn_wmma_f32_16x16x32_f16(
                      false, a, false, bfrag[kt], (short)0, acc, false, false);
        }

#pragma unroll
        for (int r = 0; r < 8; ++r) {
            float x = acc[r] + bvv;
            if (relu) x = fmaxf(x, 0.0f);
            C[(size_t)(mBase + r + 8 * hi) * ldc + oBase + n] = x;
        }
    }
}

// =====================================================================
//  Utility kernels
// =====================================================================
__global__ void fill_zero_kernel(float* __restrict__ p, int n)
{
    int i = blockIdx.x * 256 + threadIdx.x;
    if (i < n) p[i] = 0.0f;
}

// 2D convert f32 -> f16 with column placement (for K-concatenated weights)
__global__ void cvt2d_kernel(const float* __restrict__ in, int ldin,
                             _Float16* __restrict__ out, int ldout,
                             int colOff, int rows, int cols)
{
    int i = blockIdx.x * 256 + threadIdx.x;
    if (i >= rows * cols) return;
    int r = i / cols, c = i - r * cols;
    out[(size_t)r * ldout + colOff + c] = (_Float16)in[(size_t)r * ldin + c];
}

__global__ void add_vec_kernel(const float* __restrict__ a,
                               const float* __restrict__ b,
                               float* __restrict__ o, int n)
{
    int i = blockIdx.x * 256 + threadIdx.x;
    if (i < n) o[i] = a[i] + b[i];
}

// scatter-add:  dst[dstIdx[e]] += src[srcIdx[e]]   (rows of 128 floats)
// one wave per edge, float4 per lane
__global__ __launch_bounds__(256)
void edge_scatter_kernel(const float* __restrict__ src,
                         const int* __restrict__ srcIdx,
                         const int* __restrict__ dstIdx,
                         float* __restrict__ dst, int nEdges)
{
    int e = blockIdx.x * 8 + (threadIdx.x >> 5);
    if (e >= nEdges) return;
    int lane = threadIdx.x & 31;
    const float4* s = (const float4*)(src + (size_t)srcIdx[e] * H);
    float*        d = dst + (size_t)dstIdx[e] * H;
    float4 v = s[lane];
    atomicAdd(d + lane * 4 + 0, v.x);
    atomicAdd(d + lane * 4 + 1, v.y);
    atomicAdd(d + lane * 4 + 2, v.z);
    atomicAdd(d + lane * 4 + 3, v.w);
}

// LSTM pointwise: gates [N,4H] order (i,f,g,o); updates h,c in place
__global__ void lstm_pointwise_kernel(const float* __restrict__ gates,
                                      float* __restrict__ h,
                                      float* __restrict__ c, int n)
{
    int idx = blockIdx.x * 256 + threadIdx.x;
    if (idx >= n * H) return;
    int row = idx >> 7;
    int j   = idx & 127;
    const float* g = gates + (size_t)row * (4 * H);
    float gi = g[j];
    float gf = g[H + j];
    float gg = g[2 * H + j];
    float go = g[3 * H + j];
    float si = 1.0f / (1.0f + __expf(-gi));
    float sf = 1.0f / (1.0f + __expf(-gf));
    float so = 1.0f / (1.0f + __expf(-go));
    float cn = sf * c[idx] + si * tanhf(gg);
    c[idx] = cn;
    h[idx] = so * tanhf(cn);
}

// windowed multi-head attention core: one block per 16-token window
__global__ __launch_bounds__(256)
void attn_window_kernel(const float* __restrict__ qh,
                        const float* __restrict__ kh,
                        const float* __restrict__ vh,
                        float* __restrict__ o, float scale)
{
    __shared__ float sq[WWIN][H];
    __shared__ float sk[WWIN][H];
    __shared__ float sv[WWIN][H];
    __shared__ float sc[NHEAD][WWIN][WWIN];

    const size_t base = (size_t)blockIdx.x * WWIN * H;
    for (int idx = threadIdx.x; idx < WWIN * H; idx += 256) {
        int r = idx >> 7, col = idx & 127;
        sq[r][col] = qh[base + idx];
        sk[r][col] = kh[base + idx];
        sv[r][col] = vh[base + idx];
    }
    __syncthreads();

    // scores
    for (int idx = threadIdx.x; idx < NHEAD * WWIN * WWIN; idx += 256) {
        int hH = idx >> 8;
        int q  = (idx >> 4) & 15;
        int k  = idx & 15;
        float s = 0.0f;
#pragma unroll
        for (int d = 0; d < DH; ++d)
            s += sq[q][hH * DH + d] * sk[k][hH * DH + d];
        sc[hH][q][k] = s * scale;
    }
    __syncthreads();

    // softmax over k (one thread per (head, q))
    if (threadIdx.x < NHEAD * WWIN) {
        int hH = threadIdx.x >> 4;
        int q  = threadIdx.x & 15;
        float mx = -1e30f;
#pragma unroll
        for (int k = 0; k < WWIN; ++k) mx = fmaxf(mx, sc[hH][q][k]);
        float e[WWIN];
        float sum = 0.0f;
#pragma unroll
        for (int k = 0; k < WWIN; ++k) { e[k] = __expf(sc[hH][q][k] - mx); sum += e[k]; }
        float inv = 1.0f / sum;
#pragma unroll
        for (int k = 0; k < WWIN; ++k) sc[hH][q][k] = e[k] * inv;
    }
    __syncthreads();

    // o = p @ v
    for (int idx = threadIdx.x; idx < WWIN * H; idx += 256) {
        int q   = idx >> 7;
        int col = idx & 127;
        int hH  = col >> 4;
        float acc = 0.0f;
#pragma unroll
        for (int k = 0; k < WWIN; ++k) acc += sc[hH][q][k] * sv[k][col];
        o[base + idx] = acc;
    }
}

// window mean pool: out[g] = mean over 16 rows of in[g*16 .. g*16+15]
__global__ void pool16_kernel(const float* __restrict__ in,
                              float* __restrict__ out, int nOut)
{
    int idx = blockIdx.x * 256 + threadIdx.x;
    if (idx >= nOut * H) return;
    int g = idx >> 7, hcol = idx & 127;
    float s = 0.0f;
#pragma unroll
    for (int w = 0; w < WWIN; ++w)
        s += in[((size_t)g * WWIN + w) * H + hcol];
    out[idx] = s * (1.0f / 16.0f);
}

// =====================================================================
//  Host side
// =====================================================================
static inline void gemm1(hipStream_t s, const float* A,
                         const _Float16* W, const float* bias,
                         float* C, int ldc, int N, int Kout, int relu)
{
    gemm_wmma_kernel<1><<<dim3(N / 16), 256, 0, s>>>(
        A, nullptr, nullptr, 0, 0, W, bias, C, ldc, Kout, relu);
}
static inline void gemm2(hipStream_t s, const float* A0, const float* A1,
                         const _Float16* W, const float* bias,
                         float* C, int ldc, int N, int Kout, int relu)
{
    gemm_wmma_kernel<2><<<dim3(N / 16), 256, 0, s>>>(
        A0, A1, nullptr, 0, 0, W, bias, C, ldc, Kout, relu);
}
static inline void gemm3(hipStream_t s, const float* A0, const float* A1,
                         const float* A2, int shift1, int mod1,
                         const _Float16* W, const float* bias,
                         float* C, int ldc, int N, int Kout, int relu)
{
    gemm_wmma_kernel<3><<<dim3(N / 16), 256, 0, s>>>(
        A0, A1, A2, shift1, mod1, W, bias, C, ldc, Kout, relu);
}

static inline void fill0(hipStream_t s, float* p, int n)
{
    fill_zero_kernel<<<(n + 255) / 256, 256, 0, s>>>(p, n);
}

static inline void cvt2d(hipStream_t s, const float* in, int ldin,
                         _Float16* out, int ldout, int colOff,
                         int rows, int cols)
{
    cvt2d_kernel<<<(rows * cols + 255) / 256, 256, 0, s>>>(
        in, ldin, out, ldout, colOff, rows, cols);
}

extern "C" void kernel_launch(void* const* d_in, const int* in_sizes, int n_in,
                              void* d_out, int out_size, void* d_ws, size_t ws_size,
                              hipStream_t stream)
{
    (void)in_sizes; (void)n_in; (void)out_size; (void)ws_size;

    // ---- inputs ----
    const float* l_init      = (const float*)d_in[0];
    const float* c_init      = (const float*)d_in[1];
    const int*   edge_lit    = (const int*)  d_in[2];
    const int*   edge_clause = (const int*)  d_in[3];
    const float* mlp_l_w     = (const float*)d_in[4];
    const float* mlp_l_b     = (const float*)d_in[5];
    const float* mlp_c_w     = (const float*)d_in[6];
    const float* mlp_c_b     = (const float*)d_in[7];
    const float* lstm_l_wih  = (const float*)d_in[8];
    const float* lstm_l_whh  = (const float*)d_in[9];
    const float* lstm_l_bih  = (const float*)d_in[10];
    const float* lstm_l_bhh  = (const float*)d_in[11];
    const float* lstm_c_wih  = (const float*)d_in[12];
    const float* lstm_c_whh  = (const float*)d_in[13];
    const float* lstm_c_bih  = (const float*)d_in[14];
    const float* lstm_c_bhh  = (const float*)d_in[15];
    const float* wq          = (const float*)d_in[16];
    const float* bq          = (const float*)d_in[17];
    const float* wk          = (const float*)d_in[18];
    const float* bk          = (const float*)d_in[19];
    const float* wv          = (const float*)d_in[20];
    const float* bv          = (const float*)d_in[21];
    const float* attn_in_w   = (const float*)d_in[22];
    const float* attn_in_b   = (const float*)d_in[23];
    const float* attn_out_w  = (const float*)d_in[24];
    const float* attn_out_b  = (const float*)d_in[25];

    float* out = (float*)d_out;

    // ---- workspace layout ----
    char*  ws  = (char*)d_ws;
    size_t off = 0;
    auto alloc = [&](size_t bytes) -> void* {
        void* p = ws + off;
        off = (off + bytes + 255) & ~(size_t)255;
        return p;
    };
    const size_t SZ_L = (size_t)NLIT    * H * sizeof(float);   // 64 MB
    const size_t SZ_C = (size_t)NCLAUSE * H * sizeof(float);   // 32 MB

    float* lh    = (float*)alloc(SZ_L);
    float* lc    = (float*)alloc(SZ_L);
    float* ch    = (float*)alloc(SZ_C);
    float* cc    = (float*)alloc(SZ_C);
    float* bufLa = (float*)alloc(SZ_L);                 // lit tmp / lmsg / l_agg
    float* bufLb = (float*)alloc(SZ_L);                 // lit tmp
    float* bufCa = (float*)alloc(SZ_C);                 // c_in / q / o
    float* bufCb = (float*)alloc(SZ_C);                 // clause tmp / k / o2
    float* bufCc = (float*)alloc(SZ_C);                 // clause tmp / v
    float* gates = (float*)alloc((size_t)NLIT * 4 * H * sizeof(float)); // 256 MB
    float* cur1  = (float*)alloc((size_t)4096 * H * sizeof(float));
    float* cur2  = (float*)alloc((size_t)256  * H * sizeof(float));

    // f16 weights (K-concatenated where fused)
    _Float16* h_mlp_l = (_Float16*)alloc(3 * H * H * sizeof(_Float16));
    _Float16* h_mlp_c = (_Float16*)alloc(3 * H * H * sizeof(_Float16));
    _Float16* h_cgate = (_Float16*)alloc((size_t)4 * H * 2 * H * sizeof(_Float16)); // [512,256] = [wih|whh]
    _Float16* h_lgate = (_Float16*)alloc((size_t)4 * H * 3 * H * sizeof(_Float16)); // [512,384] = [wih|whh]
    _Float16* h_wq    = (_Float16*)alloc(H * H * sizeof(_Float16));
    _Float16* h_wk    = (_Float16*)alloc(H * H * sizeof(_Float16));
    _Float16* h_wv    = (_Float16*)alloc(H * H * sizeof(_Float16));
    _Float16* h_ain   = (_Float16*)alloc(LEVELS * 3 * H * H * sizeof(_Float16));
    _Float16* h_aout  = (_Float16*)alloc(LEVELS * H * H * sizeof(_Float16));
    float*    cb_c    = (float*)alloc(4 * H * sizeof(float));   // bih+bhh (clause)
    float*    cb_l    = (float*)alloc(4 * H * sizeof(float));   // bih+bhh (lit)

    // qh/kh/vh regions reuse the big gates buffer
    float* qhB = gates;
    float* khB = gates + (size_t)NCLAUSE * H;
    float* vhB = gates + (size_t)2 * NCLAUSE * H;

    // ---- init state ----
    hipMemcpyAsync(lh, l_init, SZ_L, hipMemcpyDeviceToDevice, stream);
    hipMemcpyAsync(ch, c_init, SZ_C, hipMemcpyDeviceToDevice, stream);
    fill0(stream, lc, NLIT * H);
    fill0(stream, cc, NCLAUSE * H);

    // ---- convert / concatenate weights to f16 ----
    cvt2d(stream, mlp_l_w, H, h_mlp_l, H, 0, 3 * H, H);
    cvt2d(stream, mlp_c_w, H, h_mlp_c, H, 0, 3 * H, H);
    // clause gate W' = [wih (512x128) | whh (512x128)] -> [512,256]
    cvt2d(stream, lstm_c_wih, H, h_cgate, 2 * H, 0, 4 * H, H);
    cvt2d(stream, lstm_c_whh, H, h_cgate, 2 * H, H, 4 * H, H);
    // literal gate W' = [wih (512x256) | whh (512x128)] -> [512,384]
    cvt2d(stream, lstm_l_wih, 2 * H, h_lgate, 3 * H, 0, 4 * H, 2 * H);
    cvt2d(stream, lstm_l_whh, H,     h_lgate, 3 * H, 2 * H, 4 * H, H);
    cvt2d(stream, wq, H, h_wq, H, 0, H, H);
    cvt2d(stream, wk, H, h_wk, H, 0, H, H);
    cvt2d(stream, wv, H, h_wv, H, 0, H, H);
    cvt2d(stream, attn_in_w,  H, h_ain,  H, 0, LEVELS * 3 * H, H);
    cvt2d(stream, attn_out_w, H, h_aout, H, 0, LEVELS * H, H);
    // combined biases
    add_vec_kernel<<<2, 256, 0, stream>>>(lstm_c_bih, lstm_c_bhh, cb_c, 4 * H);
    add_vec_kernel<<<2, 256, 0, stream>>>(lstm_l_bih, lstm_l_bhh, cb_l, 4 * H);

    // =============== message-passing rounds ===============
    for (int r = 0; r < NROUND; ++r) {
        // literal MLP: lmsg = L2( relu(L1( relu(L0(lh)) )) )   -> bufLa
        gemm1(stream, lh,    h_mlp_l + 0 * H * H, mlp_l_b + 0 * H, bufLa, H, NLIT, H, 1);
        gemm1(stream, bufLa, h_mlp_l + 1 * H * H, mlp_l_b + 1 * H, bufLb, H, NLIT, H, 1);
        gemm1(stream, bufLb, h_mlp_l + 2 * H * H, mlp_l_b + 2 * H, bufLa, H, NLIT, H, 0);

        // c_in = segment_sum(lmsg[edge_lit] -> edge_clause)    -> bufCa
        fill0(stream, bufCa, NCLAUSE * H);
        edge_scatter_kernel<<<E_EDGES / 8, 256, 0, stream>>>(
            bufLa, edge_lit, edge_clause, bufCa, E_EDGES);

        // clause LSTM gates = [c_in | ch] @ [wih|whh]^T + (bih+bhh)  (one pass)
        gemm2(stream, bufCa, ch, h_cgate, cb_c, gates, 4 * H, NCLAUSE, 4 * H, 0);
        lstm_pointwise_kernel<<<(NCLAUSE * H + 255) / 256, 256, 0, stream>>>(
            gates, ch, cc, NCLAUSE);

        // clause MLP: cmsg -> bufCb
        gemm1(stream, ch,    h_mlp_c + 0 * H * H, mlp_c_b + 0 * H, bufCb, H, NCLAUSE, H, 1);
        gemm1(stream, bufCb, h_mlp_c + 1 * H * H, mlp_c_b + 1 * H, bufCc, H, NCLAUSE, H, 1);
        gemm1(stream, bufCc, h_mlp_c + 2 * H * H, mlp_c_b + 2 * H, bufCb, H, NCLAUSE, H, 0);

        // l_agg = segment_sum(cmsg[edge_clause] -> edge_lit)   -> bufLa
        fill0(stream, bufLa, NLIT * H);
        edge_scatter_kernel<<<E_EDGES / 8, 256, 0, stream>>>(
            bufCb, edge_clause, edge_lit, bufLa, E_EDGES);

        // literal LSTM gates = [l_agg | lflip | lh] @ [wih|whh]^T + (bih+bhh)
        gemm3(stream, bufLa, lh, lh, NLIT / 2, NLIT, h_lgate, cb_l,
              gates, 4 * H, NLIT, 4 * H, 0);
        lstm_pointwise_kernel<<<(NLIT * H + 255) / 256, 256, 0, stream>>>(
            gates, lh, lc, NLIT);
    }

    // =============== hierarchical windowed attention ===============
    const float scale = 1.0f / 4.0f;   // 1/sqrt(DH) = 1/sqrt(16)
    float* cur  = ch;
    int    Ncur = NCLAUSE;
    for (int level = 0; level < LEVELS; ++level) {
        // q,k,v base projections
        gemm1(stream, cur, h_wq, bq, bufCa, H, Ncur, H, 0);
        gemm1(stream, cur, h_wk, bk, bufCb, H, Ncur, H, 0);
        gemm1(stream, cur, h_wv, bv, bufCc, H, Ncur, H, 0);

        // per-level head projections
        const _Float16* lw = h_ain + (size_t)level * 3 * H * H;
        const float*    lb = attn_in_b + (size_t)level * 3 * H;
        gemm1(stream, bufCa, lw + 0 * H * H, lb + 0 * H, qhB, H, Ncur, H, 0);
        gemm1(stream, bufCb, lw + 1 * H * H, lb + 1 * H, khB, H, Ncur, H, 0);
        gemm1(stream, bufCc, lw + 2 * H * H, lb + 2 * H, vhB, H, Ncur, H, 0);

        // windowed attention core -> bufCa
        attn_window_kernel<<<Ncur / WWIN, 256, 0, stream>>>(
            qhB, khB, vhB, bufCa, scale);

        // output projection -> bufCb
        gemm1(stream, bufCa, h_aout + (size_t)level * H * H,
              attn_out_b + (size_t)level * H, bufCb, H, Ncur, H, 0);

        // window mean pool
        int nNext = Ncur / WWIN;
        float* nxt = (level == LEVELS - 1) ? out
                   : ((level & 1) ? cur2 : cur1);
        pool16_kernel<<<(nNext * H + 255) / 256, 256, 0, stream>>>(
            bufCb, nxt, nNext);
        cur  = nxt;
        Ncur = nNext;
    }
    // final jnp.mean over axis 0 of [1,H] is identity: out already holds [1,128]
}
